// TemplateLayer_2516850835707
// MI455X (gfx1250) — matrix operand order repaired and checked by reading, not compiled, not run on hardware
//
#include <hip/hip_runtime.h>
#include <cstdint>
#include <cstddef>

typedef __attribute__((ext_vector_type(16))) _Float16 v16h;
typedef __attribute__((ext_vector_type(8)))  float    v8f;

#define N_FACES 200000
#define N_EDGES 300000
#define NNZ     600000
#define C_IN    128
#define C_MID   256
#define C_OUT   128

// ---------------------------------------------------------------------------
// Convert W[K,N] (f32, row-major) -> Wt[N,K] (f16, i.e. B transposed so each
// WMMA B-fragment lane reads 16 contiguous f16 along K).
// ---------------------------------------------------------------------------
__global__ void convert_transpose_w(const float* __restrict__ W,
                                    _Float16* __restrict__ Wt,
                                    int K, int N) {
    int idx = blockIdx.x * blockDim.x + threadIdx.x;
    if (idx < K * N) {
        int k = idx / N, n = idx - k * N;
        Wt[(size_t)n * K + k] = (_Float16)W[(size_t)k * N + n];
    }
}

// ---------------------------------------------------------------------------
// C[M,N] = act(A[M,K]) @ B[K,N], f32 in/out, f16 WMMA compute, f32 accum.
// One wave computes a 16 x (16*TN) output strip: TN accumulators, the A
// fragment is loaded once per K-step and reused TN times, so the large A
// operand (e.g. the 307MB edge buffer, bigger than the 192MB L2) streams
// from HBM exactly once when TN*16 == N.
//
// All TN B-fragments are loaded into a local array BEFORE the WMMA chain so
// the 2*TN b128 loads are independent: the scheduler can clause them and
// retire WMMAs against staggered s_wait_loadcnt thresholds instead of a
// load -> wait 0 -> wmma serial chain.
//
// A-fragment layout (16-bit A 16x32, contiguous K as in B): lane L holds row L&15,
//   elements 0..7  -> K = k0 + 16*(L>>4) + i
//   elements 8..15 -> K = k0 + 16*(L>>4) + i
// B-fragment layout (16-bit B 32x16): lane L holds column L&15,
//   element i -> K = k0 + 16*(L>>4) + i    (contiguous in Bt[n, :])
// C/D layout (f32 16x16): element i -> row m0 + i + 8*(L>>4), col n + (L&15)
// ---------------------------------------------------------------------------
template<int TN, bool SIG>
__global__ __launch_bounds__(256)
void wmma_gemm_f16(const float* __restrict__ A,
                   const _Float16* __restrict__ Bt,   // [N,K] f16
                   float* __restrict__ C,
                   int M, int N, int K) {
    const int lane   = threadIdx.x & 31;
    const int wid    = (blockIdx.x << 3) + (threadIdx.x >> 5);   // global wave id
    const int tilesN = N / (16 * TN);
    const int total  = (M >> 4) * tilesN;
    if (wid >= total) return;                                    // wave-uniform

    const int m0 = (wid / tilesN) << 4;
    const int n0 = (wid % tilesN) * (16 * TN);
    const int r  = lane & 15;
    const int h  = lane >> 4;

    v8f acc[TN];
#pragma unroll
    for (int t = 0; t < TN; ++t) acc[t] = (v8f){};

    const float* ap = A + (size_t)(m0 + r) * K;
    const _Float16* bp[TN];
#pragma unroll
    for (int t = 0; t < TN; ++t)
        bp[t] = Bt + (size_t)(n0 + 16 * t + r) * K + 16 * h;

    for (int k0 = 0; k0 < K; k0 += 32) {
        // A fragment: two contiguous 8-float runs per lane
        const float4* p0 = reinterpret_cast<const float4*>(ap + k0 + 16 * h);
        const float4* p1 = reinterpret_cast<const float4*>(ap + k0 + 16 * h + 8);
        float buf[16];
        *reinterpret_cast<float4*>(&buf[0])  = p0[0];
        *reinterpret_cast<float4*>(&buf[4])  = p0[1];
        *reinterpret_cast<float4*>(&buf[8])  = p1[0];
        *reinterpret_cast<float4*>(&buf[12]) = p1[1];
        if (k0 + 32 < K) __builtin_prefetch(ap + k0 + 32, 0, 3);

        // Issue all B-fragment loads up front (independent -> one clause,
        // staggered loadcnt waits, up to 2*TN loads in flight per wave).
        v16h bf[TN];
#pragma unroll
        for (int t = 0; t < TN; ++t)
            bf[t] = *reinterpret_cast<const v16h*>(bp[t] + k0);

        v16h afrag;
#pragma unroll
        for (int i = 0; i < 16; ++i) {
            float t = buf[i];
            if (SIG) t = 1.0f / (1.0f + __expf(-t));   // fused sigmoid (GEMM2)
            afrag[i] = (_Float16)t;
        }

#pragma unroll
        for (int t = 0; t < TN; ++t)
            acc[t] = __builtin_amdgcn_wmma_f32_16x16x32_f16(false, afrag, false, bf[t],
                                                            (short)0, acc[t],
                                                            false, false);
    }

    float* cp = C + (size_t)(m0 + 8 * h) * N + n0 + r;
#pragma unroll
    for (int i = 0; i < 8; ++i) {
#pragma unroll
        for (int t = 0; t < TN; ++t)
            cp[(size_t)i * N + 16 * t] = acc[t][i];
    }
}

// ---------------------------------------------------------------------------
// dst[seg[e], c] += vals[e] * src[gat[e], c]   (COO SpMM via atomics)
// one thread per (nnz, channel): lanes cover consecutive channels -> coalesced
// ---------------------------------------------------------------------------
__global__ __launch_bounds__(256)
void scatter_add_kernel(const float* __restrict__ src,
                        const int*   __restrict__ seg,
                        const int*   __restrict__ gat,
                        const float* __restrict__ vals,
                        float* __restrict__ dst, int nnz, int C) {
    long long idx = (long long)blockIdx.x * blockDim.x + threadIdx.x;
    long long total = (long long)nnz * C;
    if (idx >= total) return;
    int e = (int)(idx / C);
    int c = (int)(idx - (long long)e * C);
    int s = seg[e];
    int g = gat[e];
    atomicAdd(dst + (size_t)s * C + c, vals[e] * src[(size_t)g * C + c]);
}

__global__ __launch_bounds__(256)
void sigmoid_inplace(float* __restrict__ p, long long n) {
    long long i = (long long)blockIdx.x * blockDim.x + threadIdx.x;
    if (i < n) p[i] = 1.0f / (1.0f + __expf(-p[i]));
}

// ---------------------------------------------------------------------------
extern "C" void kernel_launch(void* const* d_in, const int* in_sizes, int n_in,
                              void* d_out, int out_size, void* d_ws, size_t ws_size,
                              hipStream_t stream) {
    const float* x        = (const float*)d_in[0];
    const int*   inc_rows = (const int*)  d_in[1];
    const int*   inc_cols = (const int*)  d_in[2];
    const float* inc_vals = (const float*)d_in[3];
    const float* W1       = (const float*)d_in[4];
    const float* W2       = (const float*)d_in[5];
    float*       out      = (float*)d_out;

    char* ws = (char*)d_ws;
    const size_t offH   = 0;                                             // h: 204.8 MB
    const size_t offE   = (size_t)N_FACES * C_MID * sizeof(float);       // e: 307.2 MB
    const size_t offW1t = offE + (size_t)N_EDGES * C_MID * sizeof(float);
    const size_t offW2t = offW1t + (size_t)C_IN * C_MID * sizeof(_Float16);

    float*    h_buf = (float*)(ws + offH);
    float*    e_buf = (float*)(ws + offE);
    float*    g_buf = (float*)(ws + offH);     // reuse: h is dead after SpMM1
    _Float16* W1t   = (_Float16*)(ws + offW1t);
    _Float16* W2t   = (_Float16*)(ws + offW2t);

    // 1) weights -> f16, transposed to [N,K]
    convert_transpose_w<<<(C_IN * C_MID + 255) / 256, 256, 0, stream>>>(W1, W1t, C_IN, C_MID);
    convert_transpose_w<<<(C_MID * C_OUT + 255) / 256, 256, 0, stream>>>(W2, W2t, C_MID, C_OUT);

    // 2) h = x @ W1        [200000,128]x[128,256], 16x128 strips (2 per row-tile)
    {
        int waves  = (N_FACES / 16) * (C_MID / 128);       // 25000
        int blocks = (waves + 7) / 8;                      // 3125
        wmma_gemm_f16<8, false><<<blocks, 256, 0, stream>>>(x, W1t, h_buf,
                                                            N_FACES, C_MID, C_IN);
    }

    // 3) e = segment_sum over edges of vals * h[cols]   (sigmoid fused into GEMM2)
    hipMemsetAsync(e_buf, 0, (size_t)N_EDGES * C_MID * sizeof(float), stream);
    {
        long long total = (long long)NNZ * C_MID;
        int blocks = (int)((total + 255) / 256);           // 600000
        scatter_add_kernel<<<blocks, 256, 0, stream>>>(h_buf, inc_rows, inc_cols,
                                                       inc_vals, e_buf, NNZ, C_MID);
    }

    // 4) g = sigmoid(e) @ W2    [300000,256]x[256,128]
    //    16x128 strips == full N, so the 307MB edge buffer (bigger than L2)
    //    streams from HBM exactly once, sigmoid computed exactly once.
    {
        int waves  = (N_EDGES / 16) * (C_OUT / 128);       // 18750
        int blocks = (waves + 7) / 8;                      // 2344
        wmma_gemm_f16<8, true><<<blocks, 256, 0, stream>>>(e_buf, W2t, g_buf,
                                                           N_EDGES, C_OUT, C_MID);
    }

    // 5) out = segment_sum over faces of vals * g[rows], then sigmoid in place
    hipMemsetAsync(out, 0, (size_t)N_FACES * C_OUT * sizeof(float), stream);
    {
        long long total = (long long)NNZ * C_OUT;
        int blocks = (int)((total + 255) / 256);           // 300000
        scatter_add_kernel<<<blocks, 256, 0, stream>>>(g_buf, inc_cols, inc_rows,
                                                       inc_vals, out, NNZ, C_OUT);
    }
    {
        long long n = (long long)N_FACES * C_OUT;          // 25.6M
        int blocks = (int)((n + 255) / 256);
        sigmoid_inplace<<<blocks, 256, 0, stream>>>(out, n);
    }
}
